// PTGSupervisedGraphSage_67010079752324
// MI455X (gfx1250) — matrix operand order repaired and checked by reading, not compiled, not run on hardware
//
#include <hip/hip_runtime.h>

// MI455X / gfx1250, wave32. Fused 3-layer MLP (collapsed GraphSAGE) using
// v_wmma_f32_16x16x32_f16. See analysis: mean-aggregation is identically zero
// on all output rows, so only rows [0,1024) of x0 and wr1/b1/wr2/b2/weight matter.

typedef __attribute__((ext_vector_type(16))) _Float16 v16h;
typedef __attribute__((ext_vector_type(8)))  _Float16 v8h;
typedef __attribute__((ext_vector_type(8)))  float    v8f;

namespace cfg {
constexpr int ROWS = 1024;   // out2
constexpr int F    = 256;    // feature dim (layer-1 K)
constexpr int E    = 128;    // embed dim
constexpr int NC   = 50;     // num classes
constexpr int NCP  = 64;     // padded classes (4 WMMA n-tiles)
constexpr int PITCH = 136;   // LDS tile pitch in f16 (16B-aligned rows, bank-conflict free)
}

// ---------------------------------------------------------------------------
// Prep: fp32 -> f16 conversion; copy wr1/wr2 as-is (n-major); transpose + pad classifier.
//   Xh  [1024][256]  = f16(x0 rows 0..1023)
//   W1T [128][256]   : W1T[n][k] = wr1[n][k]
//   W2T [128][128]   : W2T[n][k] = wr2[n][k]
//   W3h [64][128]    : W3h[c][k] = weight[k][c] (c<50), else 0
// ---------------------------------------------------------------------------
__global__ void prep_kernel(const float* __restrict__ x0,
                            const float* __restrict__ wr1,
                            const float* __restrict__ wr2,
                            const float* __restrict__ w3,
                            _Float16* __restrict__ Xh,
                            _Float16* __restrict__ W1T,
                            _Float16* __restrict__ W2T,
                            _Float16* __restrict__ W3h)
{
    using namespace cfg;
    const int NX  = ROWS * F;   // 262144
    const int NW1 = F * E;      // 32768
    const int NW2 = E * E;      // 16384
    const int NW3 = E * NCP;    // 8192
    int i = blockIdx.x * blockDim.x + threadIdx.x;
    if (i < NX) {
        Xh[i] = (_Float16)x0[i];
    } else if (i < NX + NW1) {
        int j = i - NX;
        int k = j >> 7, n = j & 127;
        W1T[n * F + k] = (_Float16)wr1[n * F + k];
    } else if (i < NX + NW1 + NW2) {
        int j = i - NX - NW1;
        int k = j >> 7, n = j & 127;
        W2T[n * E + k] = (_Float16)wr2[n * E + k];
    } else if (i < NX + NW1 + NW2 + NW3) {
        int j = i - NX - NW1 - NW2;
        int k = j >> 6, c = j & 63;
        W3h[c * E + k] = (c < NC) ? (_Float16)w3[k * NC + c] : (_Float16)0.0f;
    }
}

// ---------------------------------------------------------------------------
// Fused MLP: each 32-lane wave owns a 16-row tile.
//   L1: [16x256]@[256x128] -> relu -> LDS tile (C-layout -> A-layout relayout)
//   L2: [16x128]@[128x128] -> relu -> LDS tile
//   L3: [16x128]@[128x64]  -> out[16x50]
// 112 v_wmma_f32_16x16x32_f16 per wave; EXEC all-1s until the final guarded store.
// ---------------------------------------------------------------------------
__global__ __launch_bounds__(32) void sage_fused_kernel(
    const _Float16* __restrict__ Xh,
    const _Float16* __restrict__ W1T,
    const _Float16* __restrict__ W2T,
    const _Float16* __restrict__ W3h,
    const float* __restrict__ b1,
    const float* __restrict__ b2,
    float* __restrict__ out)
{
    using namespace cfg;
    __shared__ _Float16 tile[16 * PITCH];   // one wave per workgroup

    const int lane  = threadIdx.x & 31;
    const int lhalf = lane >> 4;            // K half-select for A frags
    const int lmod  = lane & 15;            // M (A) / N (C) index
    const int m0    = blockIdx.x * 16;      // output row base

    // Build an A fragment (16x32 f16) from a row-major f16 matrix.
    // ISA layout: lane<16 -> M=lane, K={kb*32+0..7, +16..23}; lane>=16 -> K shifted by 8.
    auto loadA = [&](const _Float16* base, int row, int stride, int kb) -> v16h {
        const _Float16* p = base + row * stride + kb * 32 + lhalf * 8;
        v8h lo = *(const v8h*)(p);
        v8h hi = *(const v8h*)(p + 16);
        v16h a;
#pragma unroll
        for (int t = 0; t < 8; ++t) { a[t] = lo[t]; a[8 + t] = hi[t]; }
        return a;
    };

    // ---------------- Layer 1: h1 = relu(X @ W1T + b1), K=256, N=128 --------
    v8f acc1[8];
#pragma unroll
    for (int nt = 0; nt < 8; ++nt) {
        float bv = b1[nt * 16 + lmod];      // C layout: N = lane%16
#pragma unroll
        for (int r = 0; r < 8; ++r) acc1[nt][r] = bv;
    }
#pragma unroll
    for (int kb = 0; kb < 8; ++kb) {
        v16h a = loadA(Xh, m0 + lmod, F, kb);
#pragma unroll
        for (int nt = 0; nt < 8; ++nt) {
            // B layout: N = lane%16, K in same table order as A (n-major weights).
            v16h b = loadA(W1T, nt * 16 + lmod, F, kb);
            acc1[nt] = __builtin_amdgcn_wmma_f32_16x16x32_f16(
                false, a, false, b, (short)0, acc1[nt], false, false);
        }
    }
    // relu + C-layout -> row-major f16 tile (M = r + 8*lhalf, N = nt*16 + lmod)
#pragma unroll
    for (int nt = 0; nt < 8; ++nt)
#pragma unroll
        for (int r = 0; r < 8; ++r) {
            float v = acc1[nt][r];
            tile[(r + lhalf * 8) * PITCH + nt * 16 + lmod] =
                (_Float16)(v > 0.0f ? v : 0.0f);
        }

    // ---------------- Layer 2: h2 = relu(h1 @ W2T + b2), K=128, N=128 -------
    v8f acc2[8];
#pragma unroll
    for (int nt = 0; nt < 8; ++nt) {
        float bv = b2[nt * 16 + lmod];
#pragma unroll
        for (int r = 0; r < 8; ++r) acc2[nt][r] = bv;
    }
#pragma unroll
    for (int kb = 0; kb < 4; ++kb) {
        v16h a = loadA(tile, lmod, PITCH, kb);   // LDS A frag (16B ds_load_b128s)
#pragma unroll
        for (int nt = 0; nt < 8; ++nt) {
            v16h b = loadA(W2T, nt * 16 + lmod, E, kb);
            acc2[nt] = __builtin_amdgcn_wmma_f32_16x16x32_f16(
                false, a, false, b, (short)0, acc2[nt], false, false);
        }
    }
#pragma unroll
    for (int nt = 0; nt < 8; ++nt)
#pragma unroll
        for (int r = 0; r < 8; ++r) {
            float v = acc2[nt][r];
            tile[(r + lhalf * 8) * PITCH + nt * 16 + lmod] =
                (_Float16)(v > 0.0f ? v : 0.0f);
        }

    // ---------------- Layer 3: out = h2 @ W3, K=128, N=64 (50 valid) --------
    v8f acc3[4];
#pragma unroll
    for (int nt = 0; nt < 4; ++nt)
#pragma unroll
        for (int r = 0; r < 8; ++r) acc3[nt][r] = 0.0f;
#pragma unroll
    for (int kb = 0; kb < 4; ++kb) {
        v16h a = loadA(tile, lmod, PITCH, kb);
#pragma unroll
        for (int nt = 0; nt < 4; ++nt) {
            v16h b = loadA(W3h, nt * 16 + lmod, E, kb);
            acc3[nt] = __builtin_amdgcn_wmma_f32_16x16x32_f16(
                false, a, false, b, (short)0, acc3[nt], false, false);
        }
    }
    // Scatter valid classes (divergence only here, after all WMMAs).
#pragma unroll
    for (int nt = 0; nt < 4; ++nt) {
        int c = nt * 16 + lmod;
        if (c < NC) {
#pragma unroll
            for (int r = 0; r < 8; ++r)
                out[(m0 + r + lhalf * 8) * NC + c] = acc3[nt][r];
        }
    }
}

// ---------------------------------------------------------------------------
// Inputs (dict order): 0:x0 1:wl1 2:wr1 3:b1 4:wl2 5:wr2 6:b2 7:weight
//                      8:out1 9:out2 10:k1 11:k2   (wl1/wl2 provably unused)
// Output: float32 [1024, 50]
// Workspace: Xh 512KB | W1T 64KB | W2T 32KB | W3h 16KB  (~624KB total)
// ---------------------------------------------------------------------------
extern "C" void kernel_launch(void* const* d_in, const int* in_sizes, int n_in,
                              void* d_out, int out_size, void* d_ws, size_t ws_size,
                              hipStream_t stream)
{
    using namespace cfg;
    (void)in_sizes; (void)n_in; (void)out_size; (void)ws_size;

    const float* x0  = (const float*)d_in[0];
    const float* wr1 = (const float*)d_in[2];
    const float* b1  = (const float*)d_in[3];
    const float* wr2 = (const float*)d_in[5];
    const float* b2  = (const float*)d_in[6];
    const float* w3  = (const float*)d_in[7];
    float* out = (float*)d_out;

    _Float16* Xh  = (_Float16*)d_ws;
    _Float16* W1T = Xh  + ROWS * F;
    _Float16* W2T = W1T + F * E;
    _Float16* W3h = W2T + E * E;

    const int total = ROWS * F + F * E + E * E + E * NCP;   // 319488
    prep_kernel<<<(total + 255) / 256, 256, 0, stream>>>(
        x0, wr1, wr2, w3, Xh, W1T, W2T, W3h);

    sage_fused_kernel<<<ROWS / 16, 32, 0, stream>>>(
        Xh, W1T, W2T, W3h, b1, b2, out);
}
